// M2_6004364280706
// MI455X (gfx1250) — compile-verified, matrix operand order checked by reading, not run on hardware
//
#include <hip/hip_runtime.h>
#include <cstdint>
#include <cstddef>

#define D_MODEL 1440
#define D_INNER 2880
#define D_STATE 128
#define HEADDIM 64
#define NHEADS  45
#define D_CONV  4
#define NLAYERS 32
#define VOCABSZ 256
#define CHUNKLEN 128
#define D_XBC   3136          // D_INNER + 2*D_STATE
#define D_PROJ  6061          // 2*D_INNER + 2*D_STATE + NHEADS
#define NP_IN   6080          // D_PROJ padded to multiple of 64
#define NP_OUT  1472          // D_MODEL padded to multiple of 64
#define BATCH   2
#define SEQLEN  2048
#define NTOK    4096
#define NCHUNK  16
#define NBLK    (BATCH*NCHUNK*NHEADS)   // 1440
#define EPSF    1e-5f

typedef __attribute__((ext_vector_type(16))) __bf16       v16bf;
typedef __attribute__((ext_vector_type(8)))  float        v8f;
typedef __attribute__((ext_vector_type(4)))  unsigned int v4u;

// ---------- helpers ----------

static __device__ __forceinline__ unsigned short f2bf(float f) {
  union { float f; unsigned u; } x; x.f = f;
  unsigned r = x.u + 0x7fffu + ((x.u >> 16) & 1u);   // round-to-nearest-even
  return (unsigned short)(r >> 16);
}

static __device__ __forceinline__ v8f vzero8() {
  v8f a;
#pragma unroll
  for (int i = 0; i < 8; ++i) a[i] = 0.f;
  return a;
}

// Load one 16x32 bf16 A/B fragment from an [R][ld] row-major bf16 array.
// Per ISA layout: lane&15 = row (M or N), lane>>4 picks K-halves; element i
// maps to K = (i>>3)*16 + (lane>>4)*8 + (i&7)  -> two contiguous b128 loads.
static __device__ __forceinline__ v16bf ldfrag(const unsigned short* p, int ld,
                                               int rowBase, int k0, int lane) {
  const unsigned short* s =
      p + (size_t)(rowBase + (lane & 15)) * ld + (k0 + ((lane >> 4) << 3));
  union { v16bf bf; v4u u[2]; } f;
  f.u[0] = *(const v4u*)(s);
  f.u[1] = *(const v4u*)(s + 16);
  return f.bf;
}

#define WMMA_BF16(a, b, c) \
  __builtin_amdgcn_wmma_f32_16x16x32_bf16(false, (a), false, (b), (short)0, (c), false, false)

// ---------- weight / activation casts ----------

// WinT[l][n][k] = Win[l][k][n], n padded to NP_IN with zeros
__global__ void m2_cast_winT(const float* __restrict__ Win, unsigned short* __restrict__ WinT) {
  long idx = (long)blockIdx.x * 256 + threadIdx.x;
  long total = (long)NLAYERS * NP_IN * D_MODEL;
  if (idx >= total) return;
  int l = (int)(idx / ((long)NP_IN * D_MODEL));
  long r = idx - (long)l * NP_IN * D_MODEL;
  int n = (int)(r / D_MODEL);
  int k = (int)(r - (long)n * D_MODEL);
  float v = (n < D_PROJ) ? Win[((size_t)l * D_MODEL + k) * D_PROJ + n] : 0.f;
  WinT[idx] = f2bf(v);
}

// WoutT[l][n][k] = Wout[l][k][n], n padded to NP_OUT
__global__ void m2_cast_woutT(const float* __restrict__ Wout, unsigned short* __restrict__ WoutT) {
  long idx = (long)blockIdx.x * 256 + threadIdx.x;
  long total = (long)NLAYERS * NP_OUT * D_INNER;
  if (idx >= total) return;
  int l = (int)(idx / ((long)NP_OUT * D_INNER));
  long r = idx - (long)l * NP_OUT * D_INNER;
  int n = (int)(r / D_INNER);
  int k = (int)(r - (long)n * D_INNER);
  float v = (n < D_MODEL) ? Wout[((size_t)l * D_INNER + k) * D_MODEL + n] : 0.f;
  WoutT[idx] = f2bf(v);
}

__global__ void m2_cast_f32_bf16(const float* __restrict__ src, unsigned short* __restrict__ dst, long n) {
  long idx = (long)blockIdx.x * 256 + threadIdx.x;
  if (idx < n) dst[idx] = f2bf(src[idx]);
}

__global__ void m2_embed(const int* __restrict__ tokens, const float* __restrict__ embed,
                         float* __restrict__ hres) {
  long idx = (long)blockIdx.x * 256 + threadIdx.x;
  if (idx >= (long)NTOK * D_MODEL) return;
  int t = (int)(idx / D_MODEL);
  int d = (int)(idx - (long)t * D_MODEL);
  hres[idx] = embed[(size_t)tokens[t] * D_MODEL + d];
}

// ---------- bf16 WMMA GEMM:  C[M][ldc] (f32) = A[M][K] (bf16) x WT[Npad][K]^T ----------
// one wave -> 32x64 output tile (2x4 grid of 16x16 WMMA tiles); 8 waves/block
__global__ __launch_bounds__(256) void m2_gemm(const unsigned short* __restrict__ A,
                                               const unsigned short* __restrict__ WT,
                                               float* __restrict__ Cm,
                                               int M, int Npad, int K, int ldc,
                                               int Nreal, int accum) {
  int lane = threadIdx.x & 31;
  int wave = threadIdx.x >> 5;
  int tilesN = Npad >> 6;
  int tile = blockIdx.x * 8 + wave;
  int tm = tile / tilesN;
  int tn = tile - tm * tilesN;
  if (tm * 32 >= M) return;
  int rowD = (lane >> 4) << 3;
  int colD = lane & 15;

  v8f acc[2][4];
#pragma unroll
  for (int i = 0; i < 2; ++i)
#pragma unroll
    for (int j = 0; j < 4; ++j) {
      if (accum) {
#pragma unroll
        for (int v = 0; v < 8; ++v) {
          int r = tm * 32 + i * 16 + rowD + v;
          int c = tn * 64 + j * 16 + colD;
          acc[i][j][v] = (c < Nreal) ? Cm[(size_t)r * ldc + c] : 0.f;
        }
      } else {
        acc[i][j] = vzero8();
      }
    }

  for (int k0 = 0; k0 < K; k0 += 32) {
    if (k0 + 32 < K)
      __builtin_prefetch(A + (size_t)(tm * 32 + (lane & 15)) * K + k0 + 32, 0, 1);
    v16bf af[2], bfv[4];
#pragma unroll
    for (int i = 0; i < 2; ++i) af[i] = ldfrag(A, K, tm * 32 + i * 16, k0, lane);
#pragma unroll
    for (int j = 0; j < 4; ++j) bfv[j] = ldfrag(WT, K, tn * 64 + j * 16, k0, lane);
#pragma unroll
    for (int i = 0; i < 2; ++i)
#pragma unroll
      for (int j = 0; j < 4; ++j)
        acc[i][j] = WMMA_BF16(af[i], bfv[j], acc[i][j]);
  }

#pragma unroll
  for (int i = 0; i < 2; ++i)
#pragma unroll
    for (int j = 0; j < 4; ++j)
#pragma unroll
      for (int v = 0; v < 8; ++v) {
        int r = tm * 32 + i * 16 + rowD + v;
        int c = tn * 64 + j * 16 + colD;
        if (c < Nreal) Cm[(size_t)r * ldc + c] = acc[i][j][v];
      }
}

// ---------- depthwise causal conv + silu ----------
__global__ void m2_conv(const float* __restrict__ zx, const float* __restrict__ cw,
                        const float* __restrict__ cb, float* __restrict__ xc) {
  long idx = (long)blockIdx.x * 256 + threadIdx.x;
  if (idx >= (long)NTOK * D_XBC) return;
  int t = (int)(idx / D_XBC);
  int c = (int)(idx - (long)t * D_XBC);
  int lpos = t & (SEQLEN - 1);
  float acc = cb[c];
#pragma unroll
  for (int j = 0; j < D_CONV; ++j) {
    int q = lpos - 3 + j;
    if (q >= 0) acc += cw[c * D_CONV + j] * zx[(size_t)(t - 3 + j) * NP_IN + D_INNER + c];
  }
  xc[idx] = acc / (1.f + __expf(-acc));   // silu
}

// ---------- dt = softplus(raw + bias) ----------
__global__ void m2_dt(const float* __restrict__ zx, const float* __restrict__ dt_bias,
                      float* __restrict__ dtb) {
  long idx = (long)blockIdx.x * 256 + threadIdx.x;
  if (idx >= (long)NTOK * NHEADS) return;
  int t = (int)(idx / NHEADS);
  int hh = (int)(idx - (long)t * NHEADS);
  float v = zx[(size_t)t * NP_IN + (D_PROJ - NHEADS) + hh] + dt_bias[hh];
  dtb[idx] = (v > 20.f) ? v : log1pf(__expf(v));
}

// ---------- SSD intra-chunk: Y_diag + chunk states (WMMA) ----------
__global__ __launch_bounds__(128) void m2_ssd1(const float* __restrict__ xc,
                                               const float* __restrict__ dtb,
                                               const float* __restrict__ A_log,
                                               float* __restrict__ ybuf,
                                               float* __restrict__ acs_ws,
                                               float* __restrict__ cstates) {
  __shared__ __align__(16) unsigned short sB[CHUNKLEN * D_STATE];    // B  [s][n]
  __shared__ __align__(16) unsigned short sBT[D_STATE * CHUNKLEN];   // B^T[n][s]
  __shared__ __align__(16) unsigned short sCM[CHUNKLEN * D_STATE];   // C, then M
  __shared__ __align__(16) unsigned short sXdT[HEADDIM * CHUNKLEN];  // Xd^T[p][s]
  __shared__ __align__(16) unsigned short sXdD[HEADDIM * CHUNKLEN];  // (Xd*dec)^T
  __shared__ float sAcs[CHUNKLEN];
  __shared__ float sDec[CHUNKLEN];

  int bid = blockIdx.x;
  int h = bid % NHEADS;
  int ch = (bid / NHEADS) % NCHUNK;
  int b = bid / (NHEADS * NCHUNK);
  int t0 = b * SEQLEN + ch * CHUNKLEN;
  int tid = threadIdx.x;
  int lane = tid & 31, wave = tid >> 5;
  float Aval = -__expf(A_log[h]);

  if (tid < CHUNKLEN) sAcs[tid] = dtb[(size_t)(t0 + tid) * NHEADS + h] * Aval;
  __syncthreads();
  if (tid == 0) {
    float s = 0.f;
    for (int i = 0; i < CHUNKLEN; ++i) { s += sAcs[i]; sAcs[i] = s; }
  }
  __syncthreads();
  float Alast = sAcs[CHUNKLEN - 1];
  if (tid < CHUNKLEN) {
    sDec[tid] = __expf(Alast - sAcs[tid]);
    acs_ws[(size_t)bid * CHUNKLEN + tid] = sAcs[tid];
  }
  __syncthreads();

  for (int idx = tid; idx < CHUNKLEN * D_STATE; idx += 128) {
    int s = idx >> 7, n = idx & 127;
    const float* row = xc + (size_t)(t0 + s) * D_XBC;
    unsigned short bv = f2bf(row[D_INNER + n]);
    sB[idx] = bv;
    sBT[n * CHUNKLEN + s] = bv;
    sCM[idx] = f2bf(row[D_INNER + D_STATE + n]);
  }
  for (int idx = tid; idx < CHUNKLEN * HEADDIM; idx += 128) {
    int s = idx >> 6, p = idx & 63;
    float dtv = dtb[(size_t)(t0 + s) * NHEADS + h];
    float v = xc[(size_t)(t0 + s) * D_XBC + h * HEADDIM + p] * dtv;
    sXdT[p * CHUNKLEN + s] = f2bf(v);
    sXdD[p * CHUNKLEN + s] = f2bf(v * sDec[s]);
  }
  __syncthreads();

  // G = C @ B^T : 8x8 tiles of 16x16, K=128. Keep in regs, then overwrite sCM with M.
  v8f g[16];
  for (int j = 0; j < 16; ++j) {
    int ti = wave * 16 + j;
    int tl = ti >> 3, ts = ti & 7;
    v8f a = vzero8();
#pragma unroll
    for (int kk = 0; kk < 4; ++kk) {
      v16bf af = ldfrag(sCM, 128, tl * 16, kk * 32, lane);
      v16bf bf = ldfrag(sB, 128, ts * 16, kk * 32, lane);
      a = WMMA_BF16(af, bf, a);
    }
    g[j] = a;
  }
  __syncthreads();
  int rowD = (lane >> 4) << 3, colD = lane & 15;
  for (int j = 0; j < 16; ++j) {
    int ti = wave * 16 + j;
    int tl = ti >> 3, ts = ti & 7;
#pragma unroll
    for (int v = 0; v < 8; ++v) {
      int l_ = tl * 16 + rowD + v;
      int s_ = ts * 16 + colD;
      float val = (l_ >= s_) ? g[j][v] * __expf(sAcs[l_] - sAcs[s_]) : 0.f;
      sCM[l_ * 128 + s_] = f2bf(val);
    }
  }
  __syncthreads();

  // Y_diag = M @ Xd : 8(l) x 4(p) tiles
  for (int j = 0; j < 8; ++j) {
    int ti = wave * 8 + j;
    int tl = ti >> 2, tp = ti & 3;
    v8f a = vzero8();
#pragma unroll
    for (int kk = 0; kk < 4; ++kk)
      a = WMMA_BF16(ldfrag(sCM, 128, tl * 16, kk * 32, lane),
                    ldfrag(sXdT, 128, tp * 16, kk * 32, lane), a);
#pragma unroll
    for (int v = 0; v < 8; ++v) {
      int l_ = tl * 16 + rowD + v;
      int p_ = tp * 16 + colD;
      ybuf[(size_t)(t0 + l_) * D_INNER + h * HEADDIM + p_] = a[v];
    }
  }

  // states = (Xd*dec)^T @ B : 4(p) x 8(n) tiles
  for (int j = 0; j < 8; ++j) {
    int ti = wave * 8 + j;
    int tp = ti >> 3, tn = ti & 7;
    v8f a = vzero8();
#pragma unroll
    for (int kk = 0; kk < 4; ++kk)
      a = WMMA_BF16(ldfrag(sXdD, 128, tp * 16, kk * 32, lane),
                    ldfrag(sBT, 128, tn * 16, kk * 32, lane), a);
#pragma unroll
    for (int v = 0; v < 8; ++v) {
      int p_ = tp * 16 + rowD + v;
      int n_ = tn * 16 + colD;
      cstates[(size_t)bid * (HEADDIM * D_STATE) + p_ * D_STATE + n_] = a[v];
    }
  }
}

// ---------- SSD inter-chunk recurrence (sequential over chunks) ----------
__global__ __launch_bounds__(256) void m2_ssd2(const float* __restrict__ cstates,
                                               const float* __restrict__ acs_ws,
                                               unsigned short* __restrict__ stin) {
  int bh = blockIdx.x;               // BATCH*NHEADS = 90
  int b = bh / NHEADS, h = bh % NHEADS;
  int tid = threadIdx.x;
  float cr[32];
#pragma unroll
  for (int i = 0; i < 32; ++i) cr[i] = 0.f;
  for (int ch = 0; ch < NCHUNK; ++ch) {
    int bid = (b * NCHUNK + ch) * NHEADS + h;
    size_t base = (size_t)bid * (HEADDIM * D_STATE);
    float dec = __expf(acs_ws[(size_t)bid * CHUNKLEN + CHUNKLEN - 1]);
#pragma unroll
    for (int i = 0; i < 32; ++i) {
      int e = tid * 32 + i;
      stin[base + e] = f2bf(cr[i]);                     // carry entering chunk
      cr[i] = cstates[base + e] + dec * cr[i];
    }
  }
}

// ---------- SSD: Y_off = exp(Acs) * (C @ carry^T) + D-skip ----------
__global__ __launch_bounds__(128) void m2_ssd3(const float* __restrict__ xc,
                                               const unsigned short* __restrict__ stin,
                                               const float* __restrict__ acs_ws,
                                               const float* __restrict__ Dp,
                                               float* __restrict__ ybuf) {
  __shared__ __align__(16) unsigned short sC[CHUNKLEN * D_STATE];
  __shared__ __align__(16) unsigned short sSt[HEADDIM * D_STATE];   // carry [p][n]
  __shared__ float sEA[CHUNKLEN];

  int bid = blockIdx.x;
  int h = bid % NHEADS;
  int ch = (bid / NHEADS) % NCHUNK;
  int b = bid / (NHEADS * NCHUNK);
  int t0 = b * SEQLEN + ch * CHUNKLEN;
  int tid = threadIdx.x;
  int lane = tid & 31, wave = tid >> 5;

  // async copy of carry tile (bf16, 16KB) global -> LDS: 128 lanes x 16B x 8 issues.
  // LDS dest address = low 32 bits of the generic pointer (wave-relative LDS offset).
  {
    unsigned lds_base = (unsigned)(uintptr_t)(void*)sSt;
    const unsigned short* g = stin + (size_t)bid * (HEADDIM * D_STATE);
#pragma unroll
    for (int it = 0; it < 8; ++it) {
      unsigned laddr = lds_base + (unsigned)((it * 2048 + tid * 8) * 2);
      unsigned long long gaddr = (unsigned long long)(uintptr_t)(g + it * 2048 + tid * 8);
      asm volatile("global_load_async_to_lds_b128 %0, %1, off"
                   :: "v"(laddr), "v"(gaddr) : "memory");
    }
  }

  if (tid < CHUNKLEN) sEA[tid] = __expf(acs_ws[(size_t)bid * CHUNKLEN + tid]);
  for (int idx = tid; idx < CHUNKLEN * D_STATE; idx += 128) {
    int s = idx >> 7, n = idx & 127;
    sC[idx] = f2bf(xc[(size_t)(t0 + s) * D_XBC + D_INNER + D_STATE + n]);
  }
  asm volatile("s_wait_asynccnt 0" ::: "memory");
  __syncthreads();

  int rowD = (lane >> 4) << 3, colD = lane & 15;
  float Dh = Dp[h];
  for (int j = 0; j < 8; ++j) {
    int ti = wave * 8 + j;
    int tl = ti >> 2, tp = ti & 3;
    v8f a = vzero8();
#pragma unroll
    for (int kk = 0; kk < 4; ++kk)
      a = WMMA_BF16(ldfrag(sC, 128, tl * 16, kk * 32, lane),
                    ldfrag(sSt, 128, tp * 16, kk * 32, lane), a);
#pragma unroll
    for (int v = 0; v < 8; ++v) {
      int l_ = tl * 16 + rowD + v;
      int p_ = tp * 16 + colD;
      int t = t0 + l_;
      int col = h * HEADDIM + p_;
      float xval = xc[(size_t)t * D_XBC + col];
      ybuf[(size_t)t * D_INNER + col] += sEA[l_] * a[v] + xval * Dh;
    }
  }
}

// ---------- gate (silu(z)) + RMSNorm -> bf16 ----------
__global__ __launch_bounds__(256) void m2_gatenorm(const float* __restrict__ zx,
                                                   const float* __restrict__ ybuf,
                                                   const float* __restrict__ nw,
                                                   unsigned short* __restrict__ yb) {
  __shared__ float sy[D_INNER];
  __shared__ float red[256];
  int t = blockIdx.x, tid = threadIdx.x;
  float loc = 0.f;
  for (int d = tid; d < D_INNER; d += 256) {
    float z = zx[(size_t)t * NP_IN + d];
    float g = ybuf[(size_t)t * D_INNER + d] * (z / (1.f + __expf(-z)));
    sy[d] = g;
    loc += g * g;
  }
  red[tid] = loc;
  __syncthreads();
  for (int off = 128; off; off >>= 1) {
    if (tid < off) red[tid] += red[tid + off];
    __syncthreads();
  }
  float inv = rsqrtf(red[0] / (float)D_INNER + EPSF);
  for (int d = tid; d < D_INNER; d += 256)
    yb[(size_t)t * D_INNER + d] = f2bf(sy[d] * inv * nw[d]);
}

// ---------- final RMSNorm -> bf16 ----------
__global__ __launch_bounds__(256) void m2_finalnorm(const float* __restrict__ hres,
                                                    const float* __restrict__ fnw,
                                                    unsigned short* __restrict__ hn) {
  __shared__ float red[256];
  int t = blockIdx.x, tid = threadIdx.x;
  float loc = 0.f;
  for (int d = tid; d < D_MODEL; d += 256) {
    float v = hres[(size_t)t * D_MODEL + d];
    loc += v * v;
  }
  red[tid] = loc;
  __syncthreads();
  for (int off = 128; off; off >>= 1) {
    if (tid < off) red[tid] += red[tid + off];
    __syncthreads();
  }
  float inv = rsqrtf(red[0] / (float)D_MODEL + EPSF);
  for (int d = tid; d < D_MODEL; d += 256)
    hn[(size_t)t * D_MODEL + d] = f2bf(hres[(size_t)t * D_MODEL + d] * inv * fnw[d]);
}

// ---------- host ----------

extern "C" void kernel_launch(void* const* d_in, const int* in_sizes, int n_in,
                              void* d_out, int out_size, void* d_ws, size_t ws_size,
                              hipStream_t stream) {
  (void)in_sizes; (void)n_in; (void)out_size; (void)ws_size;
  const int*   tokens  = (const int*)d_in[0];
  const float* embed   = (const float*)d_in[1];
  const float* Win     = (const float*)d_in[2];
  const float* conv_w  = (const float*)d_in[3];
  const float* conv_b  = (const float*)d_in[4];
  const float* dt_bias = (const float*)d_in[5];
  const float* A_log   = (const float*)d_in[6];
  const float* Dp      = (const float*)d_in[7];
  const float* norm_w  = (const float*)d_in[8];
  const float* Wout    = (const float*)d_in[9];
  const float* fnw     = (const float*)d_in[10];

  char* ws = (char*)d_ws;
  size_t off = 0;
  auto alloc = [&](size_t bytes) -> char* {
    char* p = ws + off;
    off += (bytes + 255) & ~(size_t)255;
    return p;
  };
  unsigned short* WinT  = (unsigned short*)alloc((size_t)NLAYERS * NP_IN * D_MODEL * 2);
  unsigned short* WoutT = (unsigned short*)alloc((size_t)NLAYERS * NP_OUT * D_INNER * 2);
  unsigned short* embB  = (unsigned short*)alloc((size_t)VOCABSZ * D_MODEL * 2);
  float*          hres  = (float*)alloc((size_t)NTOK * D_MODEL * 4);
  unsigned short* hb    = (unsigned short*)alloc((size_t)NTOK * D_MODEL * 2);
  unsigned short* hn    = (unsigned short*)alloc((size_t)NTOK * D_MODEL * 2);
  float*          zx    = (float*)alloc((size_t)NTOK * NP_IN * 4);
  float*          xc    = (float*)alloc((size_t)NTOK * D_XBC * 4);
  float*          dtb   = (float*)alloc((size_t)NTOK * NHEADS * 4);
  float*          ybuf  = (float*)alloc((size_t)NTOK * D_INNER * 4);
  unsigned short* yb    = (unsigned short*)alloc((size_t)NTOK * D_INNER * 2);
  float*          acs   = (float*)alloc((size_t)NBLK * CHUNKLEN * 4);
  float*          cst   = (float*)alloc((size_t)NBLK * HEADDIM * D_STATE * 4);
  unsigned short* stin  = (unsigned short*)alloc((size_t)NBLK * HEADDIM * D_STATE * 2);

  auto blocks1d = [](long n) -> unsigned { return (unsigned)((n + 255) / 256); };

  // one-time (per call) weight casts
  m2_cast_winT<<<blocks1d((long)NLAYERS * NP_IN * D_MODEL), 256, 0, stream>>>(Win, WinT);
  m2_cast_woutT<<<blocks1d((long)NLAYERS * NP_OUT * D_INNER), 256, 0, stream>>>(Wout, WoutT);
  m2_cast_f32_bf16<<<blocks1d((long)VOCABSZ * D_MODEL), 256, 0, stream>>>(embed, embB, (long)VOCABSZ * D_MODEL);
  m2_embed<<<blocks1d((long)NTOK * D_MODEL), 256, 0, stream>>>(tokens, embed, hres);

  const unsigned gridGemmIn  = (NTOK / 32) * (NP_IN / 64) / 8;   // 1520
  const unsigned gridGemmOut = (NTOK / 32) * (NP_OUT / 64) / 8;  // 368
  const unsigned gridLogits  = (NTOK / 32) * (VOCABSZ / 64) / 8; // 64

  for (int l = 0; l < NLAYERS; ++l) {
    const unsigned short* winT_l  = WinT + (size_t)l * NP_IN * D_MODEL;
    const unsigned short* woutT_l = WoutT + (size_t)l * NP_OUT * D_INNER;
    const float* cw_l  = conv_w + (size_t)l * D_XBC * D_CONV;
    const float* cb_l  = conv_b + (size_t)l * D_XBC;
    const float* dtB_l = dt_bias + (size_t)l * NHEADS;
    const float* Al_l  = A_log + (size_t)l * NHEADS;
    const float* Dp_l  = Dp + (size_t)l * NHEADS;
    const float* nw_l  = norm_w + (size_t)l * D_INNER;

    m2_cast_f32_bf16<<<blocks1d((long)NTOK * D_MODEL), 256, 0, stream>>>(hres, hb, (long)NTOK * D_MODEL);
    m2_gemm<<<gridGemmIn, 256, 0, stream>>>(hb, winT_l, zx, NTOK, NP_IN, D_MODEL, NP_IN, D_PROJ, 0);
    m2_conv<<<blocks1d((long)NTOK * D_XBC), 256, 0, stream>>>(zx, cw_l, cb_l, xc);
    m2_dt<<<blocks1d((long)NTOK * NHEADS), 256, 0, stream>>>(zx, dtB_l, dtb);
    m2_ssd1<<<NBLK, 128, 0, stream>>>(xc, dtb, Al_l, ybuf, acs, cst);
    m2_ssd2<<<BATCH * NHEADS, 256, 0, stream>>>(cst, acs, stin);
    m2_ssd3<<<NBLK, 128, 0, stream>>>(xc, stin, acs, Dp_l, ybuf);
    m2_gatenorm<<<NTOK, 256, 0, stream>>>(zx, ybuf, nw_l, yb);
    m2_gemm<<<gridGemmOut, 256, 0, stream>>>(yb, woutT_l, hres, NTOK, NP_OUT, D_INNER, D_MODEL, D_MODEL, 1);
  }

  m2_finalnorm<<<NTOK, 256, 0, stream>>>(hres, fnw, hn);
  m2_gemm<<<gridLogits, 256, 0, stream>>>(hn, embB, (float*)d_out, NTOK, VOCABSZ, D_MODEL, VOCABSZ, VOCABSZ, 0);
}
